// BigramLanguageModel_13563506721254
// MI455X (gfx1250) — compile-verified, matrix operand order checked by reading, not run on hardware
//
#include <hip/hip_runtime.h>
#include <hip/hip_bf16.h>
#include <stdint.h>
#include <math.h>

// ---- model dims ----
constexpr int kL  = 6;
constexpr int kNH = 12;
constexpr int kC  = 768;
constexpr int kHD = 64;
constexpr int kFF = 3072;
constexpr int kV  = 32000;
constexpr int kB  = 4;
constexpr int kT  = 1024;
constexpr int kBT = kB * kT;

typedef __bf16 bf16x16 __attribute__((ext_vector_type(16)));
typedef float  f32x8   __attribute__((ext_vector_type(8)));

__device__ __forceinline__ unsigned short f2bf(float f) {
  union { float f; uint32_t u; } c; c.f = f;
  uint32_t r = (c.u + 0x7FFFu + ((c.u >> 16) & 1u)) >> 16;
  return (unsigned short)r;
}

// async copy of 16B per lane: global -> LDS, tracked by ASYNCcnt.
// INST_OFFSET applies to both the LDS and global addresses (ISA 15.18.3).
__device__ __forceinline__ void async_g2l_b128(unsigned lds_off, const void* gaddr) {
  asm volatile("global_load_async_to_lds_b128 %0, %1, off"
               :: "v"(lds_off), "v"((unsigned long long)(uintptr_t)gaddr) : "memory");
}
__device__ __forceinline__ void async_g2l_b128_o16(unsigned lds_off, const void* gaddr) {
  asm volatile("global_load_async_to_lds_b128 %0, %1, off offset:16"
               :: "v"(lds_off), "v"((unsigned long long)(uintptr_t)gaddr) : "memory");
}
__device__ __forceinline__ void async_g2l_b128_o32(unsigned lds_off, const void* gaddr) {
  asm volatile("global_load_async_to_lds_b128 %0, %1, off offset:32"
               :: "v"(lds_off), "v"((unsigned long long)(uintptr_t)gaddr) : "memory");
}
__device__ __forceinline__ void async_g2l_b128_o48(unsigned lds_off, const void* gaddr) {
  asm volatile("global_load_async_to_lds_b128 %0, %1, off offset:48"
               :: "v"(lds_off), "v"((unsigned long long)(uintptr_t)gaddr) : "memory");
}
__device__ __forceinline__ void wait_async0() {
  asm volatile("s_wait_asynccnt 0x0" ::: "memory");
}

// ---------------- embedding: x = tok_emb[idx] + pos_emb ----------------
__global__ void embed_kernel(const int* __restrict__ idx,
                             const float* __restrict__ tok,
                             const float* __restrict__ pos,
                             float* __restrict__ x) {
  int m = blockIdx.x;            // token row, 0..BT-1
  int t = m % kT;
  int token = idx[m];
  const float* te = tok + (size_t)token * kC;
  const float* pe = pos + (size_t)t * kC;
  float* xr = x + (size_t)m * kC;
  for (int c = threadIdx.x; c < kC; c += blockDim.x)
    xr[c] = te[c] + pe[c];
}

// ---------------- layernorm (row=768), writes bf16 ----------------
__global__ __launch_bounds__(256) void ln_kernel(const float* __restrict__ x,
                                                 const float* __restrict__ g,
                                                 const float* __restrict__ b,
                                                 unsigned short* __restrict__ out) {
  __shared__ float red[256];
  int m = blockIdx.x;
  const float* xr = x + (size_t)m * kC;
  float lv[3];
  float s = 0.f;
  #pragma unroll
  for (int i = 0; i < 3; ++i) { lv[i] = xr[threadIdx.x + i * 256]; s += lv[i]; }
  red[threadIdx.x] = s; __syncthreads();
  for (int off = 128; off > 0; off >>= 1) {
    if (threadIdx.x < off) red[threadIdx.x] += red[threadIdx.x + off];
    __syncthreads();
  }
  float mean = red[0] * (1.0f / kC);
  __syncthreads();
  s = 0.f;
  #pragma unroll
  for (int i = 0; i < 3; ++i) { float d = lv[i] - mean; s += d * d; }
  red[threadIdx.x] = s; __syncthreads();
  for (int off = 128; off > 0; off >>= 1) {
    if (threadIdx.x < off) red[threadIdx.x] += red[threadIdx.x + off];
    __syncthreads();
  }
  float rstd = rsqrtf(red[0] * (1.0f / kC) + 1e-5f);
  unsigned short* orow = out + (size_t)m * kC;
  #pragma unroll
  for (int i = 0; i < 3; ++i) {
    int c = threadIdx.x + i * 256;
    orow[c] = f2bf((lv[i] - mean) * rstd * g[c] + b[c]);
  }
}

// ---------------- tiled bf16 WMMA GEMM, async double-buffered ----------------
// C[M,N] = alpha*A[M,K]*B + bias + resid, block tile 256x64, BK=32.
// 8 wave32s: 4 in M x 2 in N; each wave owns a 64x32 tile = 8 WMMAs/K-tile.
// A: bf16 row-major (async global->LDS). B: fp32 KxN (cvt+transpose),
// bf16 KxN (transpose), or bf16 NxK (async).
// Batch offsets: off(z) = (z/div)*s1 + (z%div)*s0.
template <bool TRANS_B, bool B_BF16, bool RELU, bool OUT_BF16>
__global__ __launch_bounds__(256) void gemm_wmma(
    int M, int N, int K, int lda, int ldb, int ldc, float alpha,
    const unsigned short* __restrict__ A, const void* __restrict__ Bp,
    void* __restrict__ Cp, const float* __restrict__ bias,
    const float* __restrict__ resid,
    int aDiv, long long aS1, long long aS0,
    int bDiv, long long bS1, long long bS0,
    int cDiv, long long cS1, long long cS0) {
  constexpr int BM = 256, BN = 64, BK = 32, LDSK = BK + 8;  // pad to 40 (80B rows)
  __shared__ unsigned short lsA[2][BM * LDSK];
  __shared__ unsigned short lsB[2][BN * LDSK];

  int z = blockIdx.z;
  A += (long long)(z / aDiv) * aS1 + (long long)(z % aDiv) * aS0;
  long long bOff = (long long)(z / bDiv) * bS1 + (long long)(z % bDiv) * bS0;
  long long cOff = (long long)(z / cDiv) * cS1 + (long long)(z % cDiv) * cS0;
  const float* Bf = (const float*)Bp + (B_BF16 ? 0 : bOff);
  const unsigned short* Bh = (const unsigned short*)Bp + (B_BF16 ? bOff : 0);
  float* Cf = (float*)Cp + (OUT_BF16 ? 0 : cOff);
  unsigned short* Chh = (unsigned short*)Cp + (OUT_BF16 ? cOff : 0);

  int tid = threadIdx.x;
  int lane = tid & 31, wave = tid >> 5;
  int wm = wave >> 1, wn = wave & 1;      // 4 waves in M x 2 waves in N
  int bm = blockIdx.y * BM, bn = blockIdx.x * BN;

  unsigned ldsAbase = (unsigned)(uintptr_t)&lsA[0][0];
  unsigned ldsBbase = (unsigned)(uintptr_t)&lsB[0][0];

  f32x8 acc[4][2] = {};
  int ktiles = K / BK;

  // ---- tile stagers ----
  auto stageA = [&](int buf, int k0) {  // 256x32 bf16: one row (64B) per thread
    const unsigned short* gp = A + (size_t)(bm + tid) * lda + k0;
    unsigned lo = ldsAbase + (unsigned)(buf * (BM * LDSK) + tid * LDSK) * 2u;
    async_g2l_b128(lo, gp);
    async_g2l_b128_o16(lo, gp);
    async_g2l_b128_o32(lo, gp);
    async_g2l_b128_o48(lo, gp);
  };
  auto stageB = [&](int buf, int k0) {
    if constexpr (TRANS_B) {            // B is N x K bf16 -> async straight copy
      int n = tid >> 2, kseg = (tid & 3) * 8;
      const unsigned short* gp = Bh + (size_t)(bn + n) * ldb + k0 + kseg;
      unsigned lo = ldsBbase + (unsigned)(buf * (BN * LDSK) + n * LDSK + kseg) * 2u;
      async_g2l_b128(lo, gp);
    } else if constexpr (B_BF16) {      // B is K x N bf16 -> transpose scatter
      int kr = tid >> 3, n0 = (tid & 7) * 8;
      uint4 raw = *(const uint4*)(Bh + (size_t)(k0 + kr) * ldb + bn + n0);
      const unsigned short* hv = (const unsigned short*)&raw;
      #pragma unroll
      for (int j = 0; j < 8; ++j) lsB[buf][(n0 + j) * LDSK + kr] = hv[j];
    } else {                            // B is K x N fp32 -> cvt bf16 + transpose
      int kr = tid >> 3, n0 = (tid & 7) * 8;
      const float* src = Bf + (size_t)(k0 + kr) * ldb + bn + n0;
      float4 r0 = *(const float4*)src;
      float4 r1 = *(const float4*)(src + 4);
      if (k0 + BK < K)
        __builtin_prefetch((const void*)(src + (size_t)BK * ldb), 0, 3);
      unsigned short hv[8] = {f2bf(r0.x), f2bf(r0.y), f2bf(r0.z), f2bf(r0.w),
                              f2bf(r1.x), f2bf(r1.y), f2bf(r1.z), f2bf(r1.w)};
      #pragma unroll
      for (int j = 0; j < 8; ++j) lsB[buf][(n0 + j) * LDSK + kr] = hv[j];
    }
  };

  // ---- software pipeline: prologue fills buffer 0 ----
  stageA(0, 0);
  stageB(0, 0);

  for (int kt = 0; kt < ktiles; ++kt) {
    int cur = kt & 1;
    wait_async0();        // this wave's async tile (buffer `cur`) has landed
    __syncthreads();      // publish to workgroup; prior reads of buf cur^1 done
    if (kt + 1 < ktiles) {
      stageA(cur ^ 1, (kt + 1) * BK);
      stageB(cur ^ 1, (kt + 1) * BK);
    }

    // ---- build fragments per ISA 16-bit A/B layout, 8 WMMAs per wave ----
    union Frag { bf16x16 v; uint32_t u[8]; };
    Frag fa[4], fb[2];
    int hl = lane >> 4, l16 = lane & 15;
    #pragma unroll
    for (int i = 0; i < 4; ++i) {
      int mr = wm * 64 + i * 16 + l16;
      const uint32_t* rowp = (const uint32_t*)&lsA[cur][mr * LDSK];
      #pragma unroll
      for (int vv = 0; vv < 8; ++vv) {
        int kp = (vv >> 2) * 16 + hl * 8 + (vv & 3) * 2;
        fa[i].u[vv] = rowp[kp >> 1];
      }
    }
    #pragma unroll
    for (int j = 0; j < 2; ++j) {
      int nr = wn * 32 + j * 16 + l16;
      const uint32_t* rowp = (const uint32_t*)&lsB[cur][nr * LDSK];
      #pragma unroll
      for (int vv = 0; vv < 8; ++vv) {
        int kp = (vv >> 2) * 16 + hl * 8 + (vv & 3) * 2;
        fb[j].u[vv] = rowp[kp >> 1];
      }
    }
    #pragma unroll
    for (int i = 0; i < 4; ++i)
      #pragma unroll
      for (int j = 0; j < 2; ++j)
        acc[i][j] = __builtin_amdgcn_wmma_f32_16x16x32_bf16(
            false, fa[i].v, false, fb[j].v, (short)0, acc[i][j], false, false);
  }

  // ---- epilogue: C/D layout row = v + 8*(lane/16), col = lane%16 ----
  int hl = lane >> 4, l16 = lane & 15;
  #pragma unroll
  for (int i = 0; i < 4; ++i) {
    #pragma unroll
    for (int j = 0; j < 2; ++j) {
      #pragma unroll
      for (int vv = 0; vv < 8; ++vv) {
        int orow = bm + wm * 64 + i * 16 + vv + 8 * hl;
        int ocol = bn + wn * 32 + j * 16 + l16;
        float val = alpha * acc[i][j][vv];
        if (bias) val += bias[ocol];
        if (RELU) val = val > 0.f ? val : 0.f;
        if (resid) val += resid[(size_t)orow * ldc + ocol];
        if (OUT_BF16) Chh[(size_t)orow * ldc + ocol] = f2bf(val);
        else          Cf [(size_t)orow * ldc + ocol] = val;
      }
    }
  }
}

// ---------------- causal softmax over T, bf16 probs out ----------------
__global__ __launch_bounds__(256) void softmax_kernel(const float* __restrict__ S,
                                                      unsigned short* __restrict__ P) {
  __shared__ float red[256];
  int row = blockIdx.x;          // over NH*B*T
  int t = row % kT;
  const float* sr = S + (size_t)row * kT;
  unsigned short* pr = P + (size_t)row * kT;
  float lv[4];
  float mx = -3.4e38f;
  #pragma unroll
  for (int i = 0; i < 4; ++i) {
    int c = threadIdx.x + i * 256;
    float v = (c <= t) ? sr[c] : -3.4e38f;
    lv[i] = v; mx = fmaxf(mx, v);
  }
  red[threadIdx.x] = mx; __syncthreads();
  for (int off = 128; off > 0; off >>= 1) {
    if (threadIdx.x < off) red[threadIdx.x] = fmaxf(red[threadIdx.x], red[threadIdx.x + off]);
    __syncthreads();
  }
  mx = red[0]; __syncthreads();
  float s = 0.f;
  #pragma unroll
  for (int i = 0; i < 4; ++i) {
    int c = threadIdx.x + i * 256;
    float e = (c <= t) ? __expf(lv[i] - mx) : 0.f;
    lv[i] = e; s += e;
  }
  red[threadIdx.x] = s; __syncthreads();
  for (int off = 128; off > 0; off >>= 1) {
    if (threadIdx.x < off) red[threadIdx.x] += red[threadIdx.x + off];
    __syncthreads();
  }
  float inv = 1.0f / red[0];
  #pragma unroll
  for (int i = 0; i < 4; ++i) {
    int c = threadIdx.x + i * 256;
    pr[c] = f2bf(lv[i] * inv);
  }
}

// ---------------- per-row NLL from logits ----------------
__global__ __launch_bounds__(256) void rowloss_kernel(const float* __restrict__ logits,
                                                      const int* __restrict__ tgt,
                                                      float* __restrict__ rl) {
  __shared__ float red[256];
  int m = blockIdx.x;
  const float* lr = logits + (size_t)m * kV;
  float mx = -3.4e38f;
  for (int c = threadIdx.x; c < kV; c += 256) mx = fmaxf(mx, lr[c]);
  red[threadIdx.x] = mx; __syncthreads();
  for (int off = 128; off > 0; off >>= 1) {
    if (threadIdx.x < off) red[threadIdx.x] = fmaxf(red[threadIdx.x], red[threadIdx.x + off]);
    __syncthreads();
  }
  mx = red[0]; __syncthreads();
  float s = 0.f;
  for (int c = threadIdx.x; c < kV; c += 256) s += __expf(lr[c] - mx);
  red[threadIdx.x] = s; __syncthreads();
  for (int off = 128; off > 0; off >>= 1) {
    if (threadIdx.x < off) red[threadIdx.x] += red[threadIdx.x + off];
    __syncthreads();
  }
  if (threadIdx.x == 0) {
    float logZ = mx + __logf(red[0]);
    rl[m] = logZ - lr[tgt[m]];
  }
}

__global__ __launch_bounds__(256) void finalloss_kernel(const float* __restrict__ rl,
                                                        float* __restrict__ out) {
  __shared__ float red[256];
  float s = 0.f;
  for (int m = threadIdx.x; m < kBT; m += 256) s += rl[m];
  red[threadIdx.x] = s; __syncthreads();
  for (int off = 128; off > 0; off >>= 1) {
    if (threadIdx.x < off) red[threadIdx.x] += red[threadIdx.x + off];
    __syncthreads();
  }
  if (threadIdx.x == 0) out[0] = red[0] / (float)kBT;
}

// ---------------- host orchestration ----------------
extern "C" void kernel_launch(void* const* d_in, const int* in_sizes, int n_in,
                              void* d_out, int out_size, void* d_ws, size_t ws_size,
                              hipStream_t stream) {
  const int*   idx     = (const int*)d_in[0];
  const int*   targets = (const int*)d_in[1];
  const float* tok     = (const float*)d_in[2];
  const float* pos     = (const float*)d_in[3];
  const float* wq      = (const float*)d_in[4];
  const float* wk      = (const float*)d_in[5];
  const float* wv      = (const float*)d_in[6];
  const float* wp      = (const float*)d_in[7];
  const float* bp      = (const float*)d_in[8];
  const float* w1      = (const float*)d_in[9];
  const float* b1      = (const float*)d_in[10];
  const float* w2      = (const float*)d_in[11];
  const float* b2      = (const float*)d_in[12];
  const float* ln1g    = (const float*)d_in[13];
  const float* ln1b    = (const float*)d_in[14];
  const float* ln2g    = (const float*)d_in[15];
  const float* ln2b    = (const float*)d_in[16];
  const float* lnfg    = (const float*)d_in[17];
  const float* lnfb    = (const float*)d_in[18];
  const float* wlm     = (const float*)d_in[19];
  const float* blm     = (const float*)d_in[20];
  (void)in_sizes; (void)n_in; (void)out_size; (void)ws_size;

  char* ws = (char*)d_ws;
  size_t off = 0;
  auto take = [&](size_t bytes) { size_t p = off; off += (bytes + 255) & ~(size_t)255; return p; };
  float*          x  = (float*)(ws + take((size_t)kBT * kC * 4));
  unsigned short* h  = (unsigned short*)(ws + take((size_t)kBT * kC * 2));
  unsigned short* q  = (unsigned short*)(ws + take((size_t)kNH * kBT * kHD * 2));
  unsigned short* kk = (unsigned short*)(ws + take((size_t)kNH * kBT * kHD * 2));
  unsigned short* v  = (unsigned short*)(ws + take((size_t)kNH * kBT * kHD * 2));
  float*          sc = (float*)(ws + take((size_t)kNH * kB * kT * kT * 4));
  unsigned short* pb = (unsigned short*)(ws + take((size_t)kNH * kB * kT * kT * 2));
  unsigned short* o  = (unsigned short*)(ws + take((size_t)kBT * kC * 2));
  unsigned short* f  = (unsigned short*)(ws + take((size_t)kBT * kFF * 2));
  float*          rl = (float*)(ws + take((size_t)kBT * 4));

  const float scale = 1.0f / sqrtf((float)kC);   // full embed dim, as in source
  float* logits = (float*)d_out;

  embed_kernel<<<kBT, 256, 0, stream>>>(idx, tok, pos, x);

  for (int l = 0; l < kL; ++l) {
    ln_kernel<<<kBT, 256, 0, stream>>>(x, ln1g + l * kC, ln1b + l * kC, h);

    // QKV: batched over heads, N=HD, per-head weight block (C x HD)
    const float* wql = wq + (size_t)l * kNH * kC * kHD;
    const float* wkl = wk + (size_t)l * kNH * kC * kHD;
    const float* wvl = wv + (size_t)l * kNH * kC * kHD;
    dim3 gq(kHD / 64, kBT / 256, kNH);
    gemm_wmma<false, false, false, true><<<gq, 256, 0, stream>>>(
        kBT, kHD, kC, kC, kHD, kHD, 1.0f, h, wql, q, nullptr, nullptr,
        1, 0, 0, 1, (long long)kC * kHD, 0, 1, (long long)kBT * kHD, 0);
    gemm_wmma<false, false, false, true><<<gq, 256, 0, stream>>>(
        kBT, kHD, kC, kC, kHD, kHD, 1.0f, h, wkl, kk, nullptr, nullptr,
        1, 0, 0, 1, (long long)kC * kHD, 0, 1, (long long)kBT * kHD, 0);
    gemm_wmma<false, false, false, true><<<gq, 256, 0, stream>>>(
        kBT, kHD, kC, kC, kHD, kHD, 1.0f, h, wvl, v, nullptr, nullptr,
        1, 0, 0, 1, (long long)kC * kHD, 0, 1, (long long)kBT * kHD, 0);

    // scores = scale * q @ k^T  : batched over (h,b), z = h*B + b
    dim3 gs(kT / 64, kT / 256, kNH * kB);
    gemm_wmma<true, true, false, false><<<gs, 256, 0, stream>>>(
        kT, kT, kHD, kHD, kHD, kT, scale, q, kk, sc, nullptr, nullptr,
        1, (long long)kT * kHD, 0, 1, (long long)kT * kHD, 0,
        1, (long long)kT * kT, 0);

    softmax_kernel<<<kNH * kB * kT, 256, 0, stream>>>(sc, pb);

    // o = P @ v, written head-concatenated: C offset = (z/B)*HD + (z%B)*T*C
    dim3 go(kHD / 64, kT / 256, kNH * kB);
    gemm_wmma<false, true, false, true><<<go, 256, 0, stream>>>(
        kT, kHD, kT, kT, kHD, kC, 1.0f, pb, v, o, nullptr, nullptr,
        1, (long long)kT * kT, 0, 1, (long long)kT * kHD, 0,
        kB, (long long)kHD, (long long)kT * kC);

    // x = x + o @ wp + bp
    dim3 gp(kC / 64, kBT / 256, 1);
    gemm_wmma<false, false, false, false><<<gp, 256, 0, stream>>>(
        kBT, kC, kC, kC, kC, kC, 1.0f, o, wp + (size_t)l * kC * kC, x,
        bp + (size_t)l * kC, x, 1, 0, 0, 1, 0, 0, 1, 0, 0);

    ln_kernel<<<kBT, 256, 0, stream>>>(x, ln2g + l * kC, ln2b + l * kC, h);

    // f = relu(h @ w1 + b1)  (bf16 out)
    dim3 g1(kFF / 64, kBT / 256, 1);
    gemm_wmma<false, false, true, true><<<g1, 256, 0, stream>>>(
        kBT, kFF, kC, kC, kFF, kFF, 1.0f, h, w1 + (size_t)l * kC * kFF, f,
        b1 + (size_t)l * kFF, nullptr, 1, 0, 0, 1, 0, 0, 1, 0, 0);

    // x = x + f @ w2 + b2
    dim3 g2(kC / 64, kBT / 256, 1);
    gemm_wmma<false, false, false, false><<<g2, 256, 0, stream>>>(
        kBT, kC, kFF, kFF, kC, kC, 1.0f, f, w2 + (size_t)l * kFF * kC, x,
        b2 + (size_t)l * kC, x, 1, 0, 0, 1, 0, 0, 1, 0, 0);
  }

  ln_kernel<<<kBT, 256, 0, stream>>>(x, lnfg, lnfb, h);

  // logits = h @ wlm + blm  -> d_out
  dim3 gl(kV / 64, kBT / 256, 1);
  gemm_wmma<false, false, false, false><<<gl, 256, 0, stream>>>(
      kBT, kV, kC, kC, kV, kV, 1.0f, h, wlm, logits, blm, nullptr,
      1, 0, 0, 1, 0, 0, 1, 0, 0);

  rowloss_kernel<<<kBT, 256, 0, stream>>>(logits, targets, rl);
  finalloss_kernel<<<1, 256, 0, stream>>>(rl, logits + (size_t)kBT * kV);
}
